// Q2A_91061896610148
// MI455X (gfx1250) — compile-verified, matrix operand order checked by reading, not run on hardware
//
#include <hip/hip_runtime.h>
#include <math.h>

#define D 768
#define DH 768
#define NHEADS 8
#define DHEAD 96
#define NB 8
#define TT 2048
#define SS 512
#define AQ 32
#define NSTEPS 16
#define LN_EPS 1e-5f

typedef __attribute__((ext_vector_type(16))) _Float16 v16h;
typedef __attribute__((ext_vector_type(8)))  float    v8f;
typedef __attribute__((ext_vector_type(4)))  float    f4;
typedef __attribute__((ext_vector_type(4)))  _Float16 h4;

// ---------------------------------------------------------------- LayerNorm (vectorized)
__global__ void ln_kernel(const float* __restrict__ x, int ldx,
                          const float* __restrict__ g, const float* __restrict__ b,
                          float* __restrict__ y, int ldy, int cols)
{
  int row = blockIdx.x;
  const f4* xr = (const f4*)(x + (size_t)row * ldx);
  f4* yr = (f4*)(y + (size_t)row * ldy);
  const f4* g4 = (const f4*)g;
  const f4* b4 = (const f4*)b;
  const int c4 = cols >> 2;
  __shared__ float red[256];
  float s = 0.f;
  for (int c = threadIdx.x; c < c4; c += 256) { f4 v = xr[c]; s += (v[0] + v[1]) + (v[2] + v[3]); }
  red[threadIdx.x] = s; __syncthreads();
  for (int o = 128; o > 0; o >>= 1) { if (threadIdx.x < o) red[threadIdx.x] += red[threadIdx.x + o]; __syncthreads(); }
  float mean = red[0] / cols;
  __syncthreads();
  float vv = 0.f;
  for (int c = threadIdx.x; c < c4; c += 256) {
    f4 v = xr[c];
#pragma unroll
    for (int e = 0; e < 4; ++e) { float d = v[e] - mean; vv += d * d; }
  }
  red[threadIdx.x] = vv; __syncthreads();
  for (int o = 128; o > 0; o >>= 1) { if (threadIdx.x < o) red[threadIdx.x] += red[threadIdx.x + o]; __syncthreads(); }
  float rstd = rsqrtf(red[0] / cols + LN_EPS);
  for (int c = threadIdx.x; c < c4; c += 256) {
    f4 v = xr[c], gg = g4[c], bb = b4[c], o;
#pragma unroll
    for (int e = 0; e < 4; ++e) o[e] = (v[e] - mean) * rstd * gg[e] + bb[e];
    yr[c] = o;
  }
}

// ---------------------------------------------------------------- Softmax (row-wise, in place)
__global__ void softmax_kernel(float* __restrict__ x, int cols)
{
  int row = blockIdx.x;
  f4* xr = (f4*)(x + (size_t)row * cols);
  const int c4 = cols >> 2;
  __shared__ float red[256];
  float m = -1e30f;
  for (int c = threadIdx.x; c < c4; c += 256) {
    f4 v = xr[c];
    m = fmaxf(m, fmaxf(fmaxf(v[0], v[1]), fmaxf(v[2], v[3])));
  }
  red[threadIdx.x] = m; __syncthreads();
  for (int o = 128; o > 0; o >>= 1) { if (threadIdx.x < o) red[threadIdx.x] = fmaxf(red[threadIdx.x], red[threadIdx.x + o]); __syncthreads(); }
  float mx = red[0]; __syncthreads();
  float s = 0.f;
  for (int c = threadIdx.x; c < c4; c += 256) {
    f4 v = xr[c];
#pragma unroll
    for (int e = 0; e < 4; ++e) { v[e] = __expf(v[e] - mx); s += v[e]; }
    xr[c] = v;
  }
  red[threadIdx.x] = s; __syncthreads();
  for (int o = 128; o > 0; o >>= 1) { if (threadIdx.x < o) red[threadIdx.x] += red[threadIdx.x + o]; __syncthreads(); }
  float inv = 1.f / red[0];
  for (int c = threadIdx.x; c < c4; c += 256) {
    f4 v = xr[c];
#pragma unroll
    for (int e = 0; e < 4; ++e) v[e] *= inv;
    xr[c] = v;
  }
}

// ---------------------------------------------------------------- WMMA GEMM
// C[M,N] = act(alpha * A[M,K] (x) B + bias[N]).   BT=true : B is [N,K] row-major (A@B^T)
//                                                 BT=false: B is [K,N] row-major (A@B)
// FULL=true: N % 128 == 0 (no bounds checks anywhere). M % 32 == 0, K % 32 == 0 required.
// 256 threads = 8 waves. Block tile 32x128; wave tile 16x32 (one A fragment, two WMMAs).
// Register double-buffered: next K-tile global loads overlap current-tile WMMAs.
template<bool BT, bool FULL>
__global__ __launch_bounds__(256) void gemm_wmma(
    const float* __restrict__ A, int lda,
    const float* __restrict__ B, int ldb,
    float* __restrict__ C, int ldc,
    int N, int K,
    const float* __restrict__ bias, float alpha, int relu)
{
  __shared__ _Float16 As[32][32];
  __shared__ _Float16 Bs[128][32];
  const int tid  = threadIdx.x;
  const int wave = tid >> 5;
  const int lane = tid & 31;
  const int wm = wave >> 2;            // 0..1
  const int wn = wave & 3;             // 0..3
  const int row0 = blockIdx.y * 32;
  const int col0 = blockIdx.x * 128;

  const int ar = tid >> 3;             // A stage: row 0..31
  const int ac = (tid & 7) << 2;       // A stage: col (float4)

  f4 ra;
  f4 rb[4];

  auto fetch = [&](int kk) {
    ra = *(const f4*)(A + (size_t)(row0 + ar) * lda + kk + ac);
    if (BT) {
#pragma unroll
      for (int j = 0; j < 4; ++j) {
        int f = tid + j * 256;
        int n = f >> 3, c = (f & 7) << 2;
        int gn = col0 + n;
        if (FULL) {
          rb[j] = *(const f4*)(B + (size_t)gn * ldb + kk + c);
        } else {
          f4 v = {0.f, 0.f, 0.f, 0.f};
          if (gn < N) v = *(const f4*)(B + (size_t)gn * ldb + kk + c);
          rb[j] = v;
        }
      }
    } else {
#pragma unroll
      for (int j = 0; j < 4; ++j) {
        int f = tid + j * 256;
        int k = f >> 5, n = (f & 31) << 2;
        int gn = col0 + n;
        if (FULL) {
          rb[j] = *(const f4*)(B + (size_t)(kk + k) * ldb + gn);
        } else {
          f4 v = {0.f, 0.f, 0.f, 0.f};
          if (gn + 3 < N) {
            v = *(const f4*)(B + (size_t)(kk + k) * ldb + gn);
          } else {
#pragma unroll
            for (int e = 0; e < 4; ++e) if (gn + e < N) v[e] = B[(size_t)(kk + k) * ldb + gn + e];
          }
          rb[j] = v;
        }
      }
    }
  };

  auto stage = [&]() {
    h4 ha = { (_Float16)ra[0], (_Float16)ra[1], (_Float16)ra[2], (_Float16)ra[3] };
    *(h4*)&As[ar][ac] = ha;
    if (BT) {
#pragma unroll
      for (int j = 0; j < 4; ++j) {
        int f = tid + j * 256;
        int n = f >> 3, c = (f & 7) << 2;
        h4 hb = { (_Float16)rb[j][0], (_Float16)rb[j][1], (_Float16)rb[j][2], (_Float16)rb[j][3] };
        *(h4*)&Bs[n][c] = hb;
      }
    } else {
#pragma unroll
      for (int j = 0; j < 4; ++j) {
        int f = tid + j * 256;
        int k = f >> 5, n = (f & 31) << 2;
#pragma unroll
        for (int e = 0; e < 4; ++e) Bs[n + e][k] = (_Float16)rb[j][e];
      }
    }
  };

  v8f acc0 = {}, acc1 = {};
  fetch(0);
  for (int kk = 0; kk < K; kk += 32) {
    stage();
    __syncthreads();
    if (kk + 64 < K)
      __builtin_prefetch(A + (size_t)(row0 + ar) * lda + kk + 64 + ac, 0, 1);  // global_prefetch_b8
    if (kk + 32 < K) fetch(kk + 32);   // overlap next-tile loads with WMMA below

    // fragments per ISA 7.12.2 (16-bit A 16x32; B mirrored, n = lane&15)
    const int m    = lane & 15;
    const int half = lane >> 4;
    const _Float16* ap = &As[wm * 16 + m][0];
    const _Float16* b0 = &Bs[wn * 16 + m][0];
    const _Float16* b1 = &Bs[64 + wn * 16 + m][0];
    v16h af, bf0, bf1;
#pragma unroll
    for (int e = 0; e < 8; ++e) {
      af[e]      = ap[half * 8 + e];
      af[8 + e]  = ap[16 + half * 8 + e];
      bf0[e]     = b0[half * 8 + e];
      bf0[8 + e] = b0[16 + half * 8 + e];
      bf1[e]     = b1[half * 8 + e];
      bf1[8 + e] = b1[16 + half * 8 + e];
    }
    acc0 = __builtin_amdgcn_wmma_f32_16x16x32_f16(false, af, false, bf0, (short)0, acc0, false, false);
    acc1 = __builtin_amdgcn_wmma_f32_16x16x32_f16(false, af, false, bf1, (short)0, acc1, false, false);
    __syncthreads();
  }

  // epilogue: lane holds col n = lane&15; acc[i] -> row i (lanes 0-15) / 8+i (lanes 16-31)
  const int nl  = wn * 16 + (lane & 15);
  const int gn0 = col0 + nl;
  const int gn1 = col0 + 64 + nl;
  const float bv0 = (bias && (FULL || gn0 < N)) ? bias[gn0] : 0.f;
  const float bv1 = (bias && (FULL || gn1 < N)) ? bias[gn1] : 0.f;
#pragma unroll
  for (int i = 0; i < 8; ++i) {
    int mrow = row0 + wm * 16 + ((lane < 16) ? i : 8 + i);
    float v0 = alpha * acc0[i] + bv0;
    float v1 = alpha * acc1[i] + bv1;
    if (relu) { v0 = fmaxf(v0, 0.f); v1 = fmaxf(v1, 0.f); }
    if (FULL || gn0 < N) C[(size_t)mrow * ldc + gn0] = v0;
    if (FULL || gn1 < N) C[(size_t)mrow * ldc + gn1] = v1;
  }
}

// ---------------------------------------------------------------- small elementwise kernels
__global__ void add_bcast_kernel(const float* __restrict__ at, const float* __restrict__ q,
                                 float* __restrict__ qa, int total)  // rows of 768, q row = r & 31
{
  int i = blockIdx.x * 256 + threadIdx.x;
  if (i < total) {
    int r = i / D, d = i - r * D;
    qa[i] = at[i] + q[(r & (AQ - 1)) * D + d];
  }
}

__global__ void copy_strided_kernel(const float* __restrict__ src, int lds,
                                    float* __restrict__ dst, int ldd, int total, int cols)
{
  int i = blockIdx.x * 256 + threadIdx.x;
  if (i < total) {
    int r = i / cols, c = i - r * cols;
    dst[(size_t)r * ldd + c] = src[(size_t)r * lds + c];
  }
}

__global__ void mask_acc_kernel(float* __restrict__ mask, const float* __restrict__ a,
                                float prev, int total)  // mask = mask*prev + a/NHEADS
{
  int i = blockIdx.x * 256 + threadIdx.x;
  if (i < total) mask[i] = mask[i] * prev + a[i] * (1.f / NHEADS);
}

__global__ void gemv_kernel(const float* __restrict__ W, const float* __restrict__ x,
                            const float* __restrict__ bias, float* __restrict__ y, int N, int K)
{
  int j = blockIdx.x * 256 + threadIdx.x;
  if (j < N) {
    float s = bias[j];
    const f4* wr = (const f4*)(W + (size_t)j * K);
    const f4* x4 = (const f4*)x;
    for (int k = 0; k < (K >> 2); ++k) {
      f4 w = wr[k], v = x4[k];
      s += (w[0] * v[0] + w[1] * v[1]) + (w[2] * v[2] + w[3] * v[3]);
    }
    y[j] = s;
  }
}

__global__ void gru_kernel(const float* __restrict__ gi,   // [AQ, 3*DH] rows of this step
                           const float* __restrict__ gh,   // [3*DH]
                           const float* __restrict__ h,    // [DH] broadcast state
                           float* __restrict__ states)     // [AQ, DH]
{
  int i = blockIdx.x * 256 + threadIdx.x;
  if (i < AQ * DH) {
    int a = i / DH, d = i - a * DH;
    const float* gr = gi + (size_t)a * 3 * DH;
    float r = 1.f / (1.f + __expf(-(gr[d] + gh[d])));
    float z = 1.f / (1.f + __expf(-(gr[DH + d] + gh[DH + d])));
    float n = tanhf(gr[2 * DH + d] + r * gh[2 * DH + d]);
    states[i] = (1.f - z) * n + z * h[d];
  }
}

// final projection (768 -> 1), argmax, state select, score write. One block of 256.
__global__ void logits_state_kernel(const float* __restrict__ xln,   // [AQ, DH] (post-LN2)
                                    const float* __restrict__ w2,    // [DH]
                                    const float* __restrict__ b2,    // [1]
                                    const float* __restrict__ states,// [AQ, DH]
                                    float* __restrict__ statev,      // [DH] (updated)
                                    float* __restrict__ out)         // [AQ]
{
  __shared__ float part[256];
  __shared__ float logits[AQ];
  __shared__ int amax;
  int a = threadIdx.x >> 3, l8 = threadIdx.x & 7;
  float s = 0.f;
  for (int k = l8; k < DH; k += 8) s += xln[(size_t)a * DH + k] * w2[k];
  part[threadIdx.x] = s; __syncthreads();
  if (l8 == 0) {
    float t = 0.f;
    for (int i = 0; i < 8; ++i) t += part[(a << 3) + i];
    logits[a] = t + b2[0];
  }
  __syncthreads();
  if (threadIdx.x == 0) {
    int best = 0; float bv = logits[0];
    for (int i = 1; i < AQ; ++i) if (logits[i] > bv) { bv = logits[i]; best = i; }
    amax = best;
  }
  __syncthreads();
  if (threadIdx.x < AQ) out[threadIdx.x] = logits[threadIdx.x];
  for (int d = threadIdx.x; d < DH; d += 256) statev[d] = states[(size_t)amax * DH + d];
}

// ---------------------------------------------------------------- host orchestration
struct MlpP { const float *b1,*b2,*l1b,*l1g,*l2b,*l2g,*w1,*w2; };

extern "C" void kernel_launch(void* const* d_in, const int* in_sizes, int n_in,
                              void* d_out, int out_size, void* d_ws, size_t ws_size,
                              hipStream_t stream)
{
  (void)in_sizes; (void)n_in; (void)out_size; (void)ws_size;
  auto F = [&](int i) -> const float* { return (const float*)d_in[i]; };

  // setup_inputs dict order, params pytree-flattened (sorted keys)
  const float* video     = F(0);
  const float* script    = F(1);
  const float* question  = F(2);
  const float* a_texts   = F(3);
  const float* a_buttons = F(4);
  const float* gru_b_hh = F(5); const float* gru_b_ih = F(6);
  const float* gru_w_hh = F(7); const float* gru_w_ih = F(8);
  MlpP mlp_pre = { F(9),  F(10), F(11), F(12), F(13), F(14), F(15), F(16) };
  MlpP mlp_t   = { F(17), F(18), F(19), F(20), F(21), F(22), F(23), F(24) };
  MlpP mlp_v   = { F(25), F(26), F(27), F(28), F(29), F(30), F(31), F(32) };
  MlpP proj    = { F(33), F(34), F(35), F(36), F(37), F(38), F(39), F(40) };
  const float* qa2s_b_in = F(41); const float* qa2s_b_out = F(42);
  const float* qa2s_w_in = F(43); const float* qa2s_w_out = F(44);
  const float* s2v_b_in  = F(45); const float* s2v_b_out  = F(46);
  const float* s2v_w_in  = F(47); const float* s2v_w_out  = F(48);
  const float* p_state   = F(49);
  float* out = (float*)d_out;

  // ---- workspace carve ----
  float* wsf = (float*)d_ws;
  size_t off = 0;
  auto alloc = [&](size_t n) { float* p = wsf + off; off += n; return p; };
  float* v_mlp  = alloc((size_t)TT * D);
  float* s_mlp  = alloc((size_t)SS * D);
  float* v_att  = alloc((size_t)SS * D);
  float* qbuf   = alloc((size_t)AQ * D);
  float* tmpA   = alloc((size_t)512 * 3072);   // == 2048*768
  float* tmpB   = alloc((size_t)512 * 3072);
  float* kh     = alloc((size_t)TT * D);
  float* vh     = alloc((size_t)TT * D);
  float* qh     = alloc((size_t)512 * D);
  float* scores = alloc((size_t)SS * TT);
  float* attno  = alloc((size_t)512 * D);
  float* khs    = alloc((size_t)SS * D);
  float* vhs    = alloc((size_t)SS * D);
  float* attnh  = alloc((size_t)512 * SS);
  float* maskb  = alloc((size_t)512 * SS);
  float* at_all = alloc((size_t)512 * D);
  float* qa_all = alloc((size_t)512 * D);
  float* inputs = alloc((size_t)512 * 4 * D);
  float* pre    = alloc((size_t)512 * D);
  float* gi     = alloc((size_t)512 * 3 * DH);
  float* gh     = alloc((size_t)3 * DH);
  float* states = alloc((size_t)AQ * DH);
  float* statev = alloc((size_t)DH);
  float* ptmp1  = alloc((size_t)AQ * DH);
  float* ptmp2  = alloc((size_t)AQ * DH);

  const float rs96 = 0.1020620726f;  // 1/sqrt(96)

  auto gemmT = [&](const float* A, int lda, const float* B, int ldb, float* C, int ldc,
                   int M, int N, int K, const float* bias, float alpha, int relu) {
    dim3 grid((N + 127) / 128, M / 32);
    if ((N & 127) == 0)
      gemm_wmma<true, true ><<<grid, 256, 0, stream>>>(A, lda, B, ldb, C, ldc, N, K, bias, alpha, relu);
    else
      gemm_wmma<true, false><<<grid, 256, 0, stream>>>(A, lda, B, ldb, C, ldc, N, K, bias, alpha, relu);
  };
  auto gemmN = [&](const float* A, int lda, const float* B, int ldb, float* C, int ldc,
                   int M, int N, int K, const float* bias, float alpha, int relu) {
    dim3 grid((N + 127) / 128, M / 32);
    if ((N & 127) == 0)
      gemm_wmma<false, true ><<<grid, 256, 0, stream>>>(A, lda, B, ldb, C, ldc, N, K, bias, alpha, relu);
    else
      gemm_wmma<false, false><<<grid, 256, 0, stream>>>(A, lda, B, ldb, C, ldc, N, K, bias, alpha, relu);
  };
  auto mlp = [&](const float* x, int rows, int din, int dout, const MlpP& p,
                 float* o, int ldout) {
    ln_kernel<<<rows, 256, 0, stream>>>(x, din, p.l1g, p.l1b, tmpA, din, din);
    gemmT(tmpA, din, p.w1, din, tmpB, din, rows, din, din, p.b1, 1.f, 1);
    ln_kernel<<<rows, 256, 0, stream>>>(tmpB, din, p.l2g, p.l2b, tmpA, din, din);
    gemmT(tmpA, din, p.w2, din, o, ldout, rows, dout, din, p.b2, 1.f, 0);
  };

  for (int b = 0; b < NB; ++b) {
    const float* vid = video     + (size_t)b * TT * D;
    const float* scr = script    + (size_t)b * SS * D;
    const float* que = question  + (size_t)b * AQ * D;
    const float* atx = a_texts   + (size_t)b * NSTEPS * AQ * D;
    const float* abt = a_buttons + (size_t)b * NSTEPS * AQ * D;

    hipMemcpyAsync(statev, p_state, DH * sizeof(float), hipMemcpyDeviceToDevice, stream);

    // ---- preamble ----
    mlp(vid, TT, D, D, mlp_v, v_mlp, D);                 // v  [T,D]
    mlp(scr, SS, D, D, mlp_t, s_mlp, D);                 // s  [S,D]
    mlp(que, AQ, D, D, mlp_t, qbuf, D);                  // q  [A,D]

    // s2v MHA (q=s, k=v=v_mlp)
    gemmT(s_mlp, D, s2v_w_in,                     D, qh, D, SS, D, D, s2v_b_in,         1.f, 0);
    gemmT(v_mlp, D, s2v_w_in + (size_t)D * D,     D, kh, D, TT, D, D, s2v_b_in + D,     1.f, 0);
    gemmT(v_mlp, D, s2v_w_in + (size_t)2 * D * D, D, vh, D, TT, D, D, s2v_b_in + 2 * D, 1.f, 0);
    for (int h = 0; h < NHEADS; ++h) {
      gemmT(qh + h * DHEAD, D, kh + h * DHEAD, D, scores, TT, SS, TT, DHEAD, nullptr, rs96, 0);
      softmax_kernel<<<SS, 256, 0, stream>>>(scores, TT);
      gemmN(scores, TT, vh + h * DHEAD, D, attno + h * DHEAD, D, SS, DHEAD, TT, nullptr, 1.f, 0);
    }
    gemmT(attno, D, s2v_w_out, D, v_att, D, SS, D, D, s2v_b_out, 1.f, 0);

    // ---- batched (state-independent) per-step work: 16*32 = 512 rows ----
    mlp(atx, NSTEPS * AQ, D, D, mlp_t, at_all, D);
    mlp(abt, NSTEPS * AQ, D, D, mlp_v, inputs + 3 * D, 4 * D);          // ab -> cols [2304,3072)
    add_bcast_kernel<<<(512 * D + 255) / 256, 256, 0, stream>>>(at_all, qbuf, qa_all, 512 * D);
    copy_strided_kernel<<<(512 * D + 255) / 256, 256, 0, stream>>>(qa_all, D, inputs + 2 * D, 4 * D, 512 * D, D);

    // qa2s MHA (q=qa rows, k=v=s) — K/V hoisted out of the scan
    gemmT(qa_all, D, qa2s_w_in,                     D, qh,  D, 512, D, D, qa2s_b_in,         1.f, 0);
    gemmT(s_mlp,  D, qa2s_w_in + (size_t)D * D,     D, khs, D, SS,  D, D, qa2s_b_in + D,     1.f, 0);
    gemmT(s_mlp,  D, qa2s_w_in + (size_t)2 * D * D, D, vhs, D, SS,  D, D, qa2s_b_in + 2 * D, 1.f, 0);
    for (int h = 0; h < NHEADS; ++h) {
      gemmT(qh + h * DHEAD, D, khs + h * DHEAD, D, attnh, SS, 512, SS, DHEAD, nullptr, rs96, 0);
      softmax_kernel<<<512, 256, 0, stream>>>(attnh, SS);
      mask_acc_kernel<<<(512 * SS + 255) / 256, 256, 0, stream>>>(maskb, attnh, (h == 0) ? 0.f : 1.f, 512 * SS);
      gemmN(attnh, SS, vhs + h * DHEAD, D, attno + h * DHEAD, D, 512, DHEAD, SS, nullptr, 1.f, 0);
    }
    gemmT(attno, D, qa2s_w_out, D, inputs + D, 4 * D, 512, D, D, qa2s_b_out, 1.f, 0);  // qa_s -> cols [768,1536)
    gemmN(maskb, SS, v_att, D, inputs, 4 * D, 512, D, SS, nullptr, 1.f, 0);            // qa_v -> cols [0,768)

    mlp(inputs, 512, 4 * D, DH, mlp_pre, pre, DH);                      // [512, DH]
    gemmT(pre, DH, gru_w_ih, DH, gi, 3 * DH, 512, 3 * DH, DH, gru_b_ih, 1.f, 0);

    // ---- serial recurrent tail ----
    for (int t = 0; t < NSTEPS; ++t) {
      gemv_kernel<<<(3 * DH + 255) / 256, 256, 0, stream>>>(gru_w_hh, statev, gru_b_hh, gh, 3 * DH, DH);
      gru_kernel<<<(AQ * DH + 255) / 256, 256, 0, stream>>>(gi + (size_t)t * AQ * 3 * DH, gh, statev, states);
      ln_kernel<<<AQ, 256, 0, stream>>>(states, DH, proj.l1g, proj.l1b, ptmp1, DH, DH);
      gemmT(ptmp1, DH, proj.w1, DH, ptmp2, DH, AQ, DH, DH, proj.b1, 1.f, 1);
      ln_kernel<<<AQ, 256, 0, stream>>>(ptmp2, DH, proj.l2g, proj.l2b, ptmp1, DH, DH);
      logits_state_kernel<<<1, 256, 0, stream>>>(ptmp1, proj.w2, proj.b2, states, statev,
                                                 out + ((size_t)b * NSTEPS + t) * AQ);
    }
  }
}
